// SAHP_27144193311522
// MI455X (gfx1250) — matrix operand order repaired and checked by reading, not compiled, not run on hardware
//
#include <hip/hip_runtime.h>

// ---------------------------------------------------------------------------
// SAHP transformer forward for MI455X (gfx1250), wave32 + WMMA bf16.
// B=2, S=1024, D=1024, H=16, DK=64, NL=4.
// All GEMM operands pre-converted to bf16; async global->LDS double-buffered
// staging; v_wmma_f32_16x16x32_bf16 compute.
// ---------------------------------------------------------------------------

#define USE_ASYNC_COPY 1   // set 0 to fall back to b128 load + ds_store staging

typedef __attribute__((ext_vector_type(16))) __bf16 v16bf;
typedef __attribute__((ext_vector_type(8)))  float  v8f;
typedef unsigned short u16;

union Frag { v16bf v; unsigned u[8]; };

static constexpr int Bc = 2;
static constexpr int Sc = 1024;
static constexpr int Dc = 1024;
static constexpr int DKc = 64;
static constexpr int Mc = Bc * Sc;          // 2048 rows

__device__ __forceinline__ u16 bf16_of(float f) {
    unsigned u = __float_as_uint(f);
    return (u16)((u + 0x7FFFu + ((u >> 16) & 1u)) >> 16);  // RNE
}
__device__ __forceinline__ unsigned pack2_bf16(float a, float b) {
    return (unsigned)bf16_of(a) | ((unsigned)bf16_of(b) << 16);
}
__device__ __forceinline__ float gelu_f(float x) {
    const float c = 0.7978845608028654f;  // sqrt(2/pi)
    float u = c * (x + 0.044715f * x * x * x);
    return 0.5f * x * (1.0f + tanhf(u));
}

// 16-byte global -> LDS copy (async on CDNA5; LDS offset = low 32 bits of
// the generic pointer, per flat-aperture mapping).
__device__ __forceinline__ void cp16(void* lds, const void* g) {
#if USE_ASYNC_COPY
    unsigned loff = (unsigned)(size_t)lds;
    asm volatile("global_load_async_to_lds_b128 %0, %1, off"
                 :: "v"(loff), "v"(g) : "memory");
#else
    *(uint4*)lds = *(const uint4*)g;
#endif
}
__device__ __forceinline__ void cp16_wait(int pending) {
#if USE_ASYNC_COPY
    if (pending) asm volatile("s_wait_asynccnt 0x4" ::: "memory");
    else         asm volatile("s_wait_asynccnt 0x0" ::: "memory");
#else
    (void)pending;
#endif
}

// ---------------------------------------------------------------------------
// Embedding + time/positional encoding (f32 out; runs once)
// ---------------------------------------------------------------------------
__global__ __launch_bounds__(256) void sahp_embed_pe(
    const float* __restrict__ dt, const int* __restrict__ types,
    const float* __restrict__ emb, const float* __restrict__ wt,
    float* __restrict__ x)
{
    const int bs = blockIdx.x;
    const int s  = bs & (Sc - 1);
    const int ty = types[bs];
    const float dtv = dt[bs];
    const float nlog = -logf(10000.0f) / (float)Dc;
    for (int d = threadIdx.x; d < Dc; d += 256) {
        const int j = (d < Dc / 2) ? d : d - Dc / 2;
        const float div = __expf(nlog * (2.0f * (float)j));
        const float ang = (float)s * div + dtv * wt[j];
        const float pe = (d < Dc / 2) ? __sinf(ang) : __cosf(ang);
        x[(size_t)bs * Dc + d] = emb[(size_t)ty * Dc + d] * 32.0f + pe;  // sqrt(1024)
    }
}

// ---------------------------------------------------------------------------
// Weight convert + transpose: W[K][N] f32 -> Wt[N][K] bf16   (one-time)
// ---------------------------------------------------------------------------
__global__ __launch_bounds__(256) void sahp_wcvt(
    const float* __restrict__ W, u16* __restrict__ Wt, int K, int N)
{
    __shared__ float tile[32][33];
    const int kb = blockIdx.y * 32, nb = blockIdx.x * 32;
    const int tx = threadIdx.x & 31, ty = threadIdx.x >> 5;   // 32 x 8
    #pragma unroll
    for (int r = 0; r < 32; r += 8)
        tile[ty + r][tx] = W[(size_t)(kb + ty + r) * N + nb + tx];
    __syncthreads();
    #pragma unroll
    for (int r = 0; r < 32; r += 8)
        Wt[(size_t)(nb + ty + r) * K + kb + tx] = bf16_of(tile[tx][ty + r]);
}

// ---------------------------------------------------------------------------
// LayerNorm (ddof=1, y = a*(x-m)/(sqrt(var)+eps)+b), bf16 out, float4 I/O
// ---------------------------------------------------------------------------
__global__ __launch_bounds__(256) void sahp_layernorm(
    const float* __restrict__ x, const float* __restrict__ a,
    const float* __restrict__ b, u16* __restrict__ y16)
{
    __shared__ float red[256];
    const int row = blockIdx.x, t = threadIdx.x;
    const float4 xv = *(const float4*)(x + (size_t)row * Dc + t * 4);
    float s = xv.x + xv.y + xv.z + xv.w;
    red[t] = s; __syncthreads();
    for (int off = 128; off > 0; off >>= 1) {
        if (t < off) red[t] += red[t + off];
        __syncthreads();
    }
    const float mean = red[0] / (float)Dc;
    __syncthreads();
    float d0 = xv.x - mean, d1 = xv.y - mean, d2 = xv.z - mean, d3 = xv.w - mean;
    red[t] = d0 * d0 + d1 * d1 + d2 * d2 + d3 * d3; __syncthreads();
    for (int off = 128; off > 0; off >>= 1) {
        if (t < off) red[t] += red[t + off];
        __syncthreads();
    }
    const float inv = 1.0f / (sqrtf(red[0] / (float)(Dc - 1)) + 1e-6f);
    const float4 av = *(const float4*)(a + t * 4);
    const float4 bv = *(const float4*)(b + t * 4);
    uint2 o;
    o.x = pack2_bf16(av.x * d0 * inv + bv.x, av.y * d1 * inv + bv.y);
    o.y = pack2_bf16(av.z * d2 * inv + bv.z, av.w * d3 * inv + bv.w);
    *(uint2*)(y16 + (size_t)row * Dc + t * 4) = o;
}

// ---------------------------------------------------------------------------
// WMMA bf16 GEMM: out = act(A16[MxK] @ Wt[NxK]^T + bias) (+res)
// Block tile 128x128, K-step 32, 8 waves as 4x2; wave tile 32x64 (8 WMMA).
// Double-buffered async LDS staging. act: 0=none,1=gelu,2=softplus(10z)/10
// ---------------------------------------------------------------------------
__global__ __launch_bounds__(256) void sahp_gemm_wmma(
    const u16* __restrict__ A16, const u16* __restrict__ Wt,
    const float* __restrict__ bias, const float* __restrict__ res,
    float* __restrict__ out, u16* __restrict__ out16, int out16_trans,
    int M, int N, int K, int act)
{
    __shared__ __align__(16) u16 As[2][128][40];
    __shared__ __align__(16) u16 Bs[2][128][40];

    const int t = threadIdx.x;
    const int lane = t & 31, wave = t >> 5;
    const int half = lane >> 4, l16 = lane & 15;
    const int wr = wave >> 1, wcl = wave & 1;        // 4 x 2 wave grid
    const int m0 = blockIdx.y * 128, n0 = blockIdx.x * 128;

    v8f acc[2][4] = {};

    // stage tile kb into buffer buf: 2x A + 2x B 16B chunks per thread
    auto stage = [&](int buf, int kb) {
        #pragma unroll
        for (int r = 0; r < 2; ++r) {
            const int id  = t + r * 256;             // 0..511
            const int row = id >> 2;                 // 0..127
            const int prt = (id & 3) * 8;            // col chunk (8 bf16 = 16B)
            cp16(&As[buf][row][prt], A16 + (size_t)(m0 + row) * K + kb + prt);
            cp16(&Bs[buf][row][prt], Wt  + (size_t)(n0 + row) * K + kb + prt);
        }
    };

    stage(0, 0);
    for (int kb = 0; kb < K; kb += 32) {
        const int buf = (kb >> 5) & 1;
        const int has_next = (kb + 32 < K);
        if (has_next) stage(buf ^ 1, kb + 32);
        cp16_wait(has_next);
        __syncthreads();                              // tile `buf` visible to all

        Frag a[2], b[4];
        #pragma unroll
        for (int f = 0; f < 2; ++f)
            #pragma unroll
            for (int i = 0; i < 8; ++i) {
                const int kp = half * 16 + i * 2;
                a[f].u[i] = *(const unsigned*)&As[buf][wr * 32 + f * 16 + l16][kp];
            }
        #pragma unroll
        for (int nb = 0; nb < 4; ++nb)
            #pragma unroll
            for (int i = 0; i < 8; ++i)
                b[nb].u[i] = *(const unsigned*)&Bs[buf][wcl * 64 + nb * 16 + l16][half * 16 + 2 * i];

        #pragma unroll
        for (int f = 0; f < 2; ++f)
            #pragma unroll
            for (int nb = 0; nb < 4; ++nb)
                acc[f][nb] = __builtin_amdgcn_wmma_f32_16x16x32_bf16(
                    false, a[f].v, false, b[nb].v, (short)0, acc[f][nb], false, false);

        __syncthreads();                              // done reading `buf`
    }

    // epilogue (C layout: row = half*8+i, col = l16)
    #pragma unroll
    for (int f = 0; f < 2; ++f)
        #pragma unroll
        for (int nb = 0; nb < 4; ++nb) {
            const int col = n0 + wcl * 64 + nb * 16 + l16;
            const float bv = bias[col];
            #pragma unroll
            for (int i = 0; i < 8; ++i) {
                const int row = m0 + wr * 32 + f * 16 + half * 8 + i;
                float v = acc[f][nb][i] + bv;
                if (act == 1) v = gelu_f(v);
                else if (act == 2) { float tt = 10.0f * v; v = (tt > 20.0f ? tt : log1pf(__expf(tt))) * 0.1f; }
                if (res) v += res[(size_t)row * N + col];
                if (out) out[(size_t)row * N + col] = v;
                if (out16)
                    out16[out16_trans ? (size_t)col * M + row : (size_t)row * N + col] = bf16_of(v);
            }
        }
}

// ---------------------------------------------------------------------------
// Flash-style causal attention, one wave per 16-query tile, bf16 q/k/vT in,
// bf16 att out. scores = exp(q.k)/sqrt(DK), tril mask, softmax, @V.
// ---------------------------------------------------------------------------
__global__ __launch_bounds__(32) void sahp_attention_wmma(
    const u16* __restrict__ q16, const u16* __restrict__ k16,
    const u16* __restrict__ v16t, u16* __restrict__ att16)
{
    __shared__ u16 Ps[16][36];

    const int lane = threadIdx.x;
    const int half = lane >> 4, l16 = lane & 15;
    const int qt  = blockIdx.x;
    const int bh  = qt >> 6;
    const int q16i = qt & 63;
    const int b   = bh >> 4, h = bh & 15;
    const int q0  = q16i * 16;
    const size_t base = (size_t)b * Sc * Dc + (size_t)h * DKc;   // q/k/att base
    const float scale = 0.125f;

    Frag qa[2];
    {
        const u16* qp = q16 + base + (size_t)(q0 + l16) * Dc;
        #pragma unroll
        for (int c = 0; c < 2; ++c)
            #pragma unroll
            for (int i = 0; i < 8; ++i) {
                const int kp = c * 32 + half * 16 + i * 2;
                qa[c].u[i] = *(const unsigned*)&qp[kp];
            }
    }

    v8f acc[4] = {};
    float mrow[8], lrow[8];
    #pragma unroll
    for (int i = 0; i < 8; ++i) { mrow[i] = -1e30f; lrow[i] = 0.0f; }

    const int kend = q0 + 16;
    for (int kb = 0; kb < kend; kb += 32) {
        v8f s0 = {}, s1 = {};
        #pragma unroll
        for (int c = 0; c < 2; ++c) {
            Frag kf0, kf1;
            const u16* kp0 = k16 + base + (size_t)(kb + l16) * Dc + c * 32 + half * 16;
            const u16* kp1 = k16 + base + (size_t)(kb + 16 + l16) * Dc + c * 32 + half * 16;
            #pragma unroll
            for (int i = 0; i < 8; ++i) {
                kf0.u[i] = *(const unsigned*)&kp0[2 * i];
                kf1.u[i] = *(const unsigned*)&kp1[2 * i];
            }
            s0 = __builtin_amdgcn_wmma_f32_16x16x32_bf16(false, qa[c].v, false, kf0.v, (short)0, s0, false, false);
            s1 = __builtin_amdgcn_wmma_f32_16x16x32_bf16(false, qa[c].v, false, kf1.v, (short)0, s1, false, false);
        }

        float pcorr[8];
        #pragma unroll
        for (int i = 0; i < 8; ++i) {
            const int srow = q0 + half * 8 + i;
            const int key0 = kb + l16, key1 = kb + 16 + l16;
            const bool ok0 = (key0 <= srow), ok1 = (key1 <= srow);
            float t0 = ok0 ? __expf(s0[i]) * scale : -1e9f;
            float t1 = ok1 ? __expf(s1[i]) * scale : -1e9f;
            float mx = fmaxf(t0, t1);
            #pragma unroll
            for (int off = 1; off < 16; off <<= 1) mx = fmaxf(mx, __shfl_xor(mx, off, 32));
            const float mnew = fmaxf(mrow[i], mx);
            const float corr = __expf(mrow[i] - mnew);
            const float p0 = ok0 ? __expf(t0 - mnew) : 0.0f;
            const float p1 = ok1 ? __expf(t1 - mnew) : 0.0f;
            float ps = p0 + p1;
            #pragma unroll
            for (int off = 1; off < 16; off <<= 1) ps += __shfl_xor(ps, off, 32);
            lrow[i] = lrow[i] * corr + ps;
            mrow[i] = mnew;
            pcorr[i] = corr;
            Ps[half * 8 + i][l16]      = bf16_of(p0);
            Ps[half * 8 + i][16 + l16] = bf16_of(p1);
        }
        #pragma unroll
        for (int nb = 0; nb < 4; ++nb)
            #pragma unroll
            for (int i = 0; i < 8; ++i) acc[nb][i] *= pcorr[i];
        __syncthreads();   // single-wave WG: dscnt fence, barrier is NOP

        Frag pa;
        #pragma unroll
        for (int i = 0; i < 8; ++i) {
            const int kp = half * 16 + i * 2;
            pa.u[i] = *(const unsigned*)&Ps[l16][kp];
        }
        #pragma unroll
        for (int nb = 0; nb < 4; ++nb) {
            Frag vb;
            const u16* vp = v16t + (size_t)(h * DKc + nb * 16 + l16) * Mc
                          + (size_t)b * Sc + kb + half * 16;
            #pragma unroll
            for (int i = 0; i < 8; ++i)
                vb.u[i] = *(const unsigned*)&vp[2 * i];
            acc[nb] = __builtin_amdgcn_wmma_f32_16x16x32_bf16(
                false, pa.v, false, vb.v, (short)0, acc[nb], false, false);
        }
        __syncthreads();
    }

    #pragma unroll
    for (int nb = 0; nb < 4; ++nb) {
        const int dk = nb * 16 + l16;
        #pragma unroll
        for (int i = 0; i < 8; ++i) {
            const int srow = q0 + half * 8 + i;
            att16[base + (size_t)srow * Dc + dk] = bf16_of(acc[nb][i] / lrow[i]);
        }
    }
}

// ---------------------------------------------------------------------------
// Host-side pipeline
// ---------------------------------------------------------------------------
extern "C" void kernel_launch(void* const* d_in, const int* in_sizes, int n_in,
                              void* d_out, int out_size, void* d_ws, size_t ws_size,
                              hipStream_t stream) {
    const float* seq_dt   = (const float*)d_in[0];
    const int*   seq_type = (const int*)  d_in[1];
    const float* emb  = (const float*)d_in[3];
    const float* wt   = (const float*)d_in[4];
    const float* wq   = (const float*)d_in[5];  const float* bq = (const float*)d_in[6];
    const float* wk   = (const float*)d_in[7];  const float* bk = (const float*)d_in[8];
    const float* wv   = (const float*)d_in[9];  const float* bv = (const float*)d_in[10];
    const float* wo   = (const float*)d_in[11]; const float* bo = (const float*)d_in[12];
    const float* w1   = (const float*)d_in[13]; const float* b1 = (const float*)d_in[14];
    const float* w2   = (const float*)d_in[15]; const float* b2 = (const float*)d_in[16];
    const float* ln1a = (const float*)d_in[17]; const float* ln1b = (const float*)d_in[18];
    const float* ln2a = (const float*)d_in[19]; const float* ln2b = (const float*)d_in[20];
    const float* wsw  = (const float*)d_in[21]; const float* bs = (const float*)d_in[22];
    const float* wc   = (const float*)d_in[23]; const float* bc = (const float*)d_in[24];
    const float* wd   = (const float*)d_in[25]; const float* bd = (const float*)d_in[26];

    const size_t T = (size_t)Mc * Dc;       // 2,097,152 elements per activation

    char* p = (char*)d_ws;
    auto alloc = [&](size_t bytes) { void* r = p; p += (bytes + 255) & ~(size_t)255; return r; };
    float* X    = (float*)alloc(T * 4);
    u16* H16    = (u16*)alloc(T * 2);
    u16* Q16    = (u16*)alloc(T * 2);
    u16* K16    = (u16*)alloc(T * 2);
    u16* V16T   = (u16*)alloc(T * 2);
    u16* ATT16  = (u16*)alloc(T * 2);
    u16* G16    = (u16*)alloc(4 * T * 2);   // 2048 x 4096
    u16* X16    = (u16*)alloc(T * 2);
    u16* WqT    = (u16*)alloc((size_t)Dc * Dc * 2);
    u16* WkT    = (u16*)alloc((size_t)Dc * Dc * 2);
    u16* WvT    = (u16*)alloc((size_t)Dc * Dc * 2);
    u16* WoT    = (u16*)alloc((size_t)Dc * Dc * 2);
    u16* W1T    = (u16*)alloc((size_t)Dc * 4 * Dc * 2);
    u16* W2T    = (u16*)alloc((size_t)Dc * 4 * Dc * 2);
    u16* WsT    = (u16*)alloc((size_t)Dc * Dc * 2);
    u16* WcT    = (u16*)alloc((size_t)Dc * Dc * 2);
    u16* WdT    = (u16*)alloc((size_t)Dc * Dc * 2);

    float* outp = (float*)d_out;

    // one-time weight convert+transpose (bf16, N-major)
    dim3 cD(Dc / 32, Dc / 32);
    sahp_wcvt<<<cD, 256, 0, stream>>>(wq, WqT, Dc, Dc);
    sahp_wcvt<<<cD, 256, 0, stream>>>(wk, WkT, Dc, Dc);
    sahp_wcvt<<<cD, 256, 0, stream>>>(wv, WvT, Dc, Dc);
    sahp_wcvt<<<cD, 256, 0, stream>>>(wo, WoT, Dc, Dc);
    sahp_wcvt<<<dim3(4 * Dc / 32, Dc / 32), 256, 0, stream>>>(w1, W1T, Dc, 4 * Dc);
    sahp_wcvt<<<dim3(Dc / 32, 4 * Dc / 32), 256, 0, stream>>>(w2, W2T, 4 * Dc, Dc);
    sahp_wcvt<<<cD, 256, 0, stream>>>(wsw, WsT, Dc, Dc);
    sahp_wcvt<<<cD, 256, 0, stream>>>(wc, WcT, Dc, Dc);
    sahp_wcvt<<<cD, 256, 0, stream>>>(wd, WdT, Dc, Dc);

    sahp_embed_pe<<<Mc, 256, 0, stream>>>(seq_dt, seq_type, emb, wt, X);

    dim3 gD(Dc / 128, Mc / 128);            // N=1024
    dim3 gF(4 * Dc / 128, Mc / 128);        // N=4096

    for (int l = 0; l < 4; ++l) {
        sahp_layernorm<<<Mc, 256, 0, stream>>>(X, ln1a, ln1b, H16);
        sahp_gemm_wmma<<<gD, 256, 0, stream>>>(H16, WqT, bq, nullptr, nullptr, Q16, 0, Mc, Dc, Dc, 0);
        sahp_gemm_wmma<<<gD, 256, 0, stream>>>(H16, WkT, bk, nullptr, nullptr, K16, 0, Mc, Dc, Dc, 0);
        sahp_gemm_wmma<<<gD, 256, 0, stream>>>(H16, WvT, bv, nullptr, nullptr, V16T, 1, Mc, Dc, Dc, 0);
        sahp_attention_wmma<<<Bc * 16 * (Sc / 16), 32, 0, stream>>>(Q16, K16, V16T, ATT16);
        sahp_gemm_wmma<<<gD, 256, 0, stream>>>(ATT16, WoT, bo, X, X, nullptr, 0, Mc, Dc, Dc, 0);
        sahp_layernorm<<<Mc, 256, 0, stream>>>(X, ln2a, ln2b, H16);
        sahp_gemm_wmma<<<gF, 256, 0, stream>>>(H16, W1T, b1, nullptr, nullptr, G16, 0, Mc, 4 * Dc, Dc, 1);
        sahp_gemm_wmma<<<gD, 256, 0, stream>>>(G16, W2T, b2, X, X, X16, 0, Mc, Dc, 4 * Dc, 0);
    }

    sahp_gemm_wmma<<<gD, 256, 0, stream>>>(X16, WsT, bs, nullptr, outp,         nullptr, 0, Mc, Dc, Dc, 1);
    sahp_gemm_wmma<<<gD, 256, 0, stream>>>(X16, WcT, bc, nullptr, outp + 1 * T, nullptr, 0, Mc, Dc, Dc, 1);
    sahp_gemm_wmma<<<gD, 256, 0, stream>>>(X16, WdT, bd, nullptr, outp + 2 * T, nullptr, 0, Mc, Dc, Dc, 2);
}
